// STN_28793460752885
// MI455X (gfx1250) — matrix operand order repaired and checked by reading, not compiled, not run on hardware
//
#include <hip/hip_runtime.h>

typedef __attribute__((ext_vector_type(16))) _Float16 v16h;
typedef __attribute__((ext_vector_type(8)))  float    v8f;

union AFrag { v16h v; _Float16 h[16]; };

__device__ __forceinline__ int clampi(int v, int lo, int hi) {
    return v < lo ? lo : (v > hi ? hi : v);
}

#define THREADS 256

// Async copy 16 bytes global -> LDS (CDNA5 ASYNCcnt path, per-lane addressing).
__device__ __forceinline__ void async_copy_b128(const void* g, unsigned lds_off) {
    asm volatile("global_load_async_to_lds_b128 %0, %1, off"
                 :: "v"(lds_off), "v"(g) : "memory");
}
__device__ __forceinline__ void wait_async0() {
    asm volatile("s_wait_asynccnt 0" ::: "memory");
}

// Pack f32 HWIO weights into f16 per-chunk WMMA-B fragment order:
// wp[((c*(CO/16)+tg)*512) + Ln*16 + hh], k = 32c + (hh>=8?16:0)+(Ln>=16?8:0)+(hh&7), n = tg*16+(Ln&15)
__global__ __launch_bounds__(THREADS)
void pack_weights(const float* __restrict__ w, _Float16* __restrict__ wp,
                  int K, int CO, int total)
{
    int idx = blockIdx.x * blockDim.x + threadIdx.x;
    if (idx >= total) return;
    int e  = idx & 511;
    int g9 = idx >> 9;
    int ntg = CO >> 4;
    int tg = g9 % ntg;
    int c  = g9 / ntg;
    int Ln = e >> 4, hh = e & 15;
    int k = c * 32 + ((hh >= 8) ? 16 : 0) + ((Ln >= 16) ? 8 : 0) + (hh & 7);
    int n = tg * 16 + (Ln & 15);
    float v = (k < K) ? w[k * CO + n] : 0.0f;
    wp[idx] = (_Float16)v;
}

// Implicit-GEMM 3x3 VALID conv + bias + ReLU on v_wmma_f32_16x16x32_f16.
// Block = 8 waves; wave owns a 16(M) x 16*NT(N) tile. Prepacked f16 weight
// chunks are async-DMA'd to LDS (double buffered, one wait+barrier per chunk).
// All dims compile-time so A-loads become base + immediate-offset b128s.
template<int NT, int CI, int CO, int HI, int WI>
__global__ __launch_bounds__(THREADS)
void conv3x3_relu_wmma(const float* __restrict__ x, const _Float16* __restrict__ wp,
                       const float* __restrict__ bias, float* __restrict__ y)
{
    constexpr int HO = HI - 2, WO = WI - 2;
    constexpr int K  = 9 * CI;
    constexpr int NCHUNK = (K + 31) / 32;
    constexpr int M  = 32 * HO * WO;          // always a multiple of 16
    constexpr int NTG = CO / 16;              // N tiles over full CO
    constexpr int CHUNK_HALF = NT * 512;      // f16 elems staged per chunk

    __shared__ __align__(16) _Float16 Bs[2][CHUNK_HALF];

    const int tid   = threadIdx.x;
    const int lane  = tid & 31;
    const int wave  = tid >> 5;
    const int mlane = lane & 15;
    const int tileM = (blockIdx.x * 8 + wave) * 16;
    const int tileN = blockIdx.y * (16 * NT);
    const bool live = (tileM < M);            // wave-uniform

    // m -> (b, oy, ox); dead waves clamp (reads stay in bounds, stores skipped)
    int mc = tileM + mlane;
    if (mc >= M) mc = M - 1;
    const int b  = mc / (HO * WO);
    const int r  = mc - b * (HO * WO);
    const int oy = r / WO;
    const int ox = r - oy * WO;
    const float* __restrict__ xbase = x + ((b * HI + oy) * WI + ox) * CI;

    const int kA = (lane >= 16) ? 8 : 0;      // A-frag K base (ISA 16-bit A layout)

    // Stage one prepacked chunk (NT*1KB) into Bs[buf] via async DMA to LDS.
    auto stageB = [&](int c, int buf) {
        if (tid < CHUNK_HALF / 8) {           // 16B per thread
            const void* g = wp + (c * NTG + (tileN >> 4)) * 512 + tid * 8;
            unsigned lds_off = (unsigned)(size_t)(&Bs[buf][tid * 8]);
            async_copy_b128(g, lds_off);
        }
    };

    // Per-lane im2col A fragment (16 halves), f32 -> f16.
    auto loadA = [&](int c, AFrag& a) {
        if constexpr (CI % 16 == 0) {
            // 8-half group g: one (ky,kx), 8 consecutive ci; kA only offsets ci.
            #pragma unroll
            for (int g = 0; g < 2; ++g) {
                const int ks0 = c * 32 + 16 * g;     // compile-time after unroll
                const int e   = ks0 / CI;
                const int ci0 = ks0 - e * CI;
                const int ky  = e / 3;
                const int kx  = e - ky * 3;
                const float* p = xbase + (ky * WI + kx) * CI + ci0 + kA;
                float4 v0 = *reinterpret_cast<const float4*>(p);
                float4 v1 = *reinterpret_cast<const float4*>(p + 4);
                a.h[8 * g + 0] = (_Float16)v0.x;
                a.h[8 * g + 1] = (_Float16)v0.y;
                a.h[8 * g + 2] = (_Float16)v0.z;
                a.h[8 * g + 3] = (_Float16)v0.w;
                a.h[8 * g + 4] = (_Float16)v1.x;
                a.h[8 * g + 5] = (_Float16)v1.y;
                a.h[8 * g + 6] = (_Float16)v1.z;
                a.h[8 * g + 7] = (_Float16)v1.w;
            }
        } else {
            // CI==3 (K=27, single chunk): rows are 3 contiguous 9-float runs.
            float xv[27];
            #pragma unroll
            for (int row = 0; row < 3; ++row) {
                const float* p = xbase + row * (WI * CI);
                #pragma unroll
                for (int q = 0; q < 9; ++q) xv[row * 9 + q] = p[q];
            }
            #pragma unroll
            for (int hh = 0; hh < 16; ++hh) {
                const int kL = hh + ((hh >= 8) ? 8 : 0);  // lanes 0-15
                const int kH = kL + 8;                    // lanes 16-31
                float vL = xv[kL];
                float vH = (kH < 27) ? xv[kH] : 0.0f;
                a.h[hh] = (_Float16)((lane < 16) ? vL : vH);
            }
        }
    };

    v8f acc[NT];
    const v8f vzero = {};
    #pragma unroll
    for (int t = 0; t < NT; ++t) acc[t] = vzero;

    stageB(0, 0);
    wait_async0();
    __syncthreads();

    #pragma unroll
    for (int c = 0; c < NCHUNK; ++c) {
        AFrag a;
        loadA(c, a);                                   // A loads in flight
        if (c + 1 < NCHUNK) stageB(c + 1, (c + 1) & 1);
        const _Float16* bp = &Bs[c & 1][lane << 4];
        #pragma unroll
        for (int t = 0; t < NT; ++t) {
            v16h bf = *reinterpret_cast<const v16h*>(bp + t * 512);
            acc[t] = __builtin_amdgcn_wmma_f32_16x16x32_f16(false, a.v, false, bf,
                                                            (short)0, acc[t],
                                                            false, false);
        }
        if (c + 1 < NCHUNK) {
            wait_async0();
            __syncthreads();
        }
    }

    // Epilogue: C/D layout — VGPR g: lanes 0-15 row M=g, lanes 16-31 row M=8+g.
    if (live) {
        const int rbase = (lane >= 16) ? 8 : 0;
        float* yb = y + (tileM + rbase) * CO + tileN + mlane;
        #pragma unroll
        for (int t = 0; t < NT; ++t) {
            const float bn = bias[tileN + t * 16 + mlane];
            #pragma unroll
            for (int g = 0; g < 8; ++g) {
                float v = acc[t][g] + bn;
                yb[g * CO + t * 16] = v > 0.0f ? v : 0.0f;   // immediate offsets
            }
        }
    }
}

// 2x2/2 max pool, 4 channels per thread (C % 4 == 0 always here).
__global__ __launch_bounds__(THREADS)
void maxpool2x2_v4(const float4* __restrict__ in, float4* __restrict__ out,
                   int Hi, int Wi, int C4, int Ho, int Wo, int total4)
{
    int idx = blockIdx.x * blockDim.x + threadIdx.x;
    if (idx >= total4) return;
    int c = idx % C4;
    int t = idx / C4;
    int ow = t % Wo; t /= Wo;
    int oh = t % Ho;
    int bb = t / Ho;
    int base = ((bb * Hi + 2 * oh) * Wi + 2 * ow) * C4 + c;
    int rs = Wi * C4;
    float4 a = in[base], b4 = in[base + C4];
    float4 cc = in[base + rs], d = in[base + rs + C4];
    float4 o;
    o.x = fmaxf(fmaxf(a.x, b4.x), fmaxf(cc.x, d.x));
    o.y = fmaxf(fmaxf(a.y, b4.y), fmaxf(cc.y, d.y));
    o.z = fmaxf(fmaxf(a.z, b4.z), fmaxf(cc.z, d.z));
    o.w = fmaxf(fmaxf(a.w, b4.w), fmaxf(cc.w, d.w));
    out[idx] = o;
}

// Global average pool over HW positions, 4 channels per thread.
__global__ __launch_bounds__(THREADS)
void gap_v4(const float4* __restrict__ in, float4* __restrict__ out, int HW, int C4)
{
    int idx = blockIdx.x * blockDim.x + threadIdx.x;
    if (idx >= 32 * C4) return;
    int b = idx / C4, c = idx - b * C4;
    const float4* p = in + b * HW * C4 + c;
    float4 s = {0.0f, 0.0f, 0.0f, 0.0f};
    for (int i = 0; i < HW; ++i) {
        float4 v = p[i * C4];
        s.x += v.x; s.y += v.y; s.z += v.z; s.w += v.w;
    }
    float inv = 1.0f / (float)HW;
    s.x *= inv; s.y *= inv; s.z *= inv; s.w *= inv;
    out[idx] = s;
}

// feat(32x128) -> relu(@D1 128x128) -> relu(@D2 128x64) -> @D3 64x6 + db3 = theta(32x6)
__global__ __launch_bounds__(THREADS)
void dense_head(const float* __restrict__ feat,
                const float* __restrict__ D1, const float* __restrict__ db1,
                const float* __restrict__ D2, const float* __restrict__ db2,
                const float* __restrict__ D3, const float* __restrict__ db3,
                float* __restrict__ theta)
{
    __shared__ float sfeat[32 * 128];
    __shared__ float sh1[32 * 128];
    __shared__ float sh2[32 * 64];
    const int tid = threadIdx.x;
    for (int i = tid; i < 32 * 128; i += THREADS) sfeat[i] = feat[i];
    __syncthreads();
    for (int i = tid; i < 32 * 128; i += THREADS) {
        int b = i >> 7, j = i & 127;
        float acc = db1[j];
        for (int k = 0; k < 128; ++k) acc += sfeat[(b << 7) + k] * D1[(k << 7) + j];
        sh1[i] = acc > 0.0f ? acc : 0.0f;
    }
    __syncthreads();
    for (int i = tid; i < 32 * 64; i += THREADS) {
        int b = i >> 6, j = i & 63;
        float acc = db2[j];
        for (int k = 0; k < 128; ++k) acc += sh1[(b << 7) + k] * D2[(k << 6) + j];
        sh2[i] = acc > 0.0f ? acc : 0.0f;
    }
    __syncthreads();
    for (int i = tid; i < 32 * 6; i += THREADS) {
        int b = i / 6, j = i - b * 6;
        float acc = db3[j];
        for (int k = 0; k < 64; ++k) acc += sh2[(b << 6) + k] * D3[k * 6 + j];
        theta[i] = acc;
    }
}

// Affine grid + bilinear sampling of original x (32,224,224,3)
__global__ __launch_bounds__(THREADS)
void stn_sampler(const float* __restrict__ x, const float* __restrict__ theta,
                 float* __restrict__ out, int total)
{
    int idx = blockIdx.x * blockDim.x + threadIdx.x;
    if (idx >= total) return;
    const int HW = 224 * 224;
    int b = idx / HW;
    int p = idx - b * HW;
    int h = p / 224;
    int wcol = p - h * 224;
    const float* t = theta + b * 6;
    float xs = -1.0f + 2.0f * (float)wcol * (1.0f / 223.0f);
    float ys = -1.0f + 2.0f * (float)h    * (1.0f / 223.0f);
    float xg = t[0] * xs + t[1] * ys + t[2];
    float yg = t[3] * xs + t[4] * ys + t[5];
    float xf = 0.5f * (xg + 1.0f) * 222.0f;
    float yf = 0.5f * (yg + 1.0f) * 222.0f;
    int x0 = (int)floorf(xf);
    int y0 = (int)floorf(yf);
    int x1 = clampi(x0 + 1, 0, 223);
    int y1 = clampi(y0 + 1, 0, 223);
    x0 = clampi(x0, 0, 223);
    y0 = clampi(y0, 0, 223);
    float x0f = (float)x0, x1f = (float)x1, y0f = (float)y0, y1f = (float)y1;
    float wa = (x1f - xf) * (y1f - yf);
    float wb = (x1f - xf) * (yf - y0f);
    float wc = (xf - x0f) * (y1f - yf);
    float wd = (xf - x0f) * (yf - y0f);
    const float* base = x + b * HW * 3;
    const float* Ia = base + (y0 * 224 + x0) * 3;
    const float* Ib = base + (y1 * 224 + x0) * 3;
    const float* Ic = base + (y0 * 224 + x1) * 3;
    const float* Id = base + (y1 * 224 + x1) * 3;
    float* o = out + idx * 3;
    #pragma unroll
    for (int c = 0; c < 3; ++c)
        o[c] = wa * Ia[c] + wb * Ib[c] + wc * Ic[c] + wd * Id[c];
}

extern "C" void kernel_launch(void* const* d_in, const int* in_sizes, int n_in,
                              void* d_out, int out_size, void* d_ws, size_t ws_size,
                              hipStream_t stream)
{
    const float* x   = (const float*)d_in[0];
    const float* W1  = (const float*)d_in[1];
    const float* b1  = (const float*)d_in[2];
    const float* W2  = (const float*)d_in[3];
    const float* b2  = (const float*)d_in[4];
    const float* W3  = (const float*)d_in[5];
    const float* b3  = (const float*)d_in[6];
    const float* D1  = (const float*)d_in[7];
    const float* db1 = (const float*)d_in[8];
    const float* D2  = (const float*)d_in[9];
    const float* db2 = (const float*)d_in[10];
    const float* D3  = (const float*)d_in[11];
    const float* db3 = (const float*)d_in[12];
    float* out = (float*)d_out;
    float* ws  = (float*)d_ws;

    // Aliased workspace: big region holds conv outputs, A-region holds pool outputs.
    float* reg0  = ws;                      // 50,466,816 f32 (conv1/2/3 out, reused)
    float* regA  = ws + 50466816L;          // 12,616,704 f32 (pool1/2/3 out, reused)
    float* feat  = ws + 63083520L;          // 4096 f32
    float* theta = feat + 4096;             // 192 f32
    _Float16* wp1 = (_Float16*)(ws + 63087808L);  // 1024 f16 (16B aligned)
    _Float16* wp2 = wp1 + 1024;                   // 18432 f16
    _Float16* wp3 = wp2 + 18432;                  // 73728 f16

    // Prepack weights into f16 fragment-order chunks (tiny, once per call).
    pack_weights<<<(1024  + THREADS - 1) / THREADS, THREADS, 0, stream>>>(W1, wp1, 27, 32, 1024);
    pack_weights<<<(18432 + THREADS - 1) / THREADS, THREADS, 0, stream>>>(W2, wp2, 288, 64, 18432);
    pack_weights<<<(73728 + THREADS - 1) / THREADS, THREADS, 0, stream>>>(W3, wp3, 576, 128, 73728);

    // conv1: (32,224,224,3) -> (32,222,222,32)  NT=2 covers all 32 N
    {
        constexpr int M = 32 * 222 * 222;
        dim3 g((M / 16 + 7) / 8, 1);
        conv3x3_relu_wmma<2, 3, 32, 224, 224><<<g, THREADS, 0, stream>>>(x, wp1, b1, reg0);
    }
    // pool1 -> (32,111,111,32)
    {
        int total4 = 32 * 111 * 111 * 8;
        maxpool2x2_v4<<<(total4 + THREADS - 1) / THREADS, THREADS, 0, stream>>>(
            (const float4*)reg0, (float4*)regA, 222, 222, 8, 111, 111, total4);
    }
    // conv2: (32,111,111,32) -> (32,109,109,64)  NT=4 covers all 64 N
    {
        constexpr int M = 32 * 109 * 109;
        dim3 g((M / 16 + 7) / 8, 1);
        conv3x3_relu_wmma<4, 32, 64, 111, 111><<<g, THREADS, 0, stream>>>(regA, wp2, b2, reg0);
    }
    // pool2 -> (32,54,54,64)
    {
        int total4 = 32 * 54 * 54 * 16;
        maxpool2x2_v4<<<(total4 + THREADS - 1) / THREADS, THREADS, 0, stream>>>(
            (const float4*)reg0, (float4*)regA, 109, 109, 16, 54, 54, total4);
    }
    // conv3: (32,54,54,64) -> (32,52,52,128)  NT=4, grid.y = 2
    {
        constexpr int M = 32 * 52 * 52;
        dim3 g((M / 16 + 7) / 8, 2);
        conv3x3_relu_wmma<4, 64, 128, 54, 54><<<g, THREADS, 0, stream>>>(regA, wp3, b3, reg0);
    }
    // pool3 -> (32,26,26,128)
    {
        int total4 = 32 * 26 * 26 * 32;
        maxpool2x2_v4<<<(total4 + THREADS - 1) / THREADS, THREADS, 0, stream>>>(
            (const float4*)reg0, (float4*)regA, 52, 52, 32, 26, 26, total4);
    }
    // global average pool -> feat(32,128)
    gap_v4<<<4, THREADS, 0, stream>>>((const float4*)regA, (float4*)feat, 26 * 26, 32);
    // MLP head -> theta(32,6)
    dense_head<<<1, THREADS, 0, stream>>>(feat, D1, db1, D2, db2, D3, db3, theta);
    // affine grid + bilinear sampler -> out(32,224,224,3)
    {
        int total = 32 * 224 * 224;
        stn_sampler<<<(total + THREADS - 1) / THREADS, THREADS, 0, stream>>>(
            x, theta, out, total);
    }
}
